// S5Encoder_44203803410968
// MI455X (gfx1250) — matrix operand order repaired and checked by reading, not compiled, not run on hardware
//
#include <hip/hip_runtime.h>
#include <math.h>

typedef __bf16 v16bf __attribute__((ext_vector_type(16)));
typedef float  v8f   __attribute__((ext_vector_type(8)));

namespace {

constexpr int LDIM = 8192;
constexpr int HDIM = 1024;
constexpr int PDIM = 1024;
constexpr int CHUNK  = 128;
constexpr int NCHUNK = LDIM / CHUNK;   // 64
constexpr float EPS = 1e-5f;

__device__ __forceinline__ unsigned short f2bf(float f) {
  union { float f; unsigned int u; } c; c.f = f;
  unsigned int u = c.u;
  return (unsigned short)((u + 0x7FFFu + ((u >> 16) & 1u)) >> 16);
}

union Frag { v16bf v; unsigned int u[8]; };

// 16-bit A-matrix 16x32 layout: lane m=lane%16, khigh=lane/16,
// VGPR v holds K = {2v, 2v+1} offset by khigh*16 (contiguous, matches B).
__device__ __forceinline__ void load_frag_a(const unsigned short* s, int row, int khigh, Frag& f) {
  const unsigned short* rp = s + row * 32 + khigh * 16;
#pragma unroll
  for (int v = 0; v < 8; ++v) {
    int kbase = 2 * v;
    f.u[v] = *(const unsigned int*)(rp + kbase);
  }
}

// 16-bit B-matrix 32x16 layout: lane n=lane%16; lanes 0-15 hold K=0..15,
// lanes 16-31 hold K=16..31, contiguous across the 8 VGPRs (2 halves each).
__device__ __forceinline__ void load_frag_b(const unsigned short* s, int row, int khigh, Frag& f) {
  const uint4* rp = (const uint4*)(s + row * 32 + khigh * 16);
  uint4 q0 = rp[0], q1 = rp[1];
  f.u[0] = q0.x; f.u[1] = q0.y; f.u[2] = q0.z; f.u[3] = q0.w;
  f.u[4] = q1.x; f.u[5] = q1.y; f.u[6] = q1.z; f.u[7] = q1.w;
}

// CDNA5 async global->LDS copy (ASYNCcnt path, no VGPR round-trip).
// VDST VGPR carries the per-lane LDS byte address; flat shared-aperture
// addresses have the group-segment offset in the low 32 bits.
__device__ __forceinline__ void async_b128(const void* g, const void* l) {
  unsigned lds = (unsigned)(unsigned long long)l;
  asm volatile("global_load_async_to_lds_b128 %0, %1, off"
               :: "v"(lds), "v"(g)
               : "memory");
}

template <int N>
__device__ __forceinline__ void wait_async() {
  asm volatile("s_wait_asynccnt %0" :: "i"(N) : "memory");
}

} // namespace

// ---------------- setup: discretization scalars per state p ----------------
__global__ void s5_setup(const float* __restrict__ lre, const float* __restrict__ lim,
                         const float* __restrict__ logdt,
                         float* __restrict__ abr, float* __restrict__ abi,
                         float* __restrict__ apr, float* __restrict__ api,
                         float* __restrict__ cfr, float* __restrict__ cfi) {
  int p = blockIdx.x * blockDim.x + threadIdx.x;
  if (p >= PDIM) return;
  float dt = expf(logdt[p]);
  float lr = lre[p], li = lim[p];
  float zr = lr * dt, zi = li * dt;
  float s, c;
  float er = expf(zr);
  sincosf(zi, &s, &c);
  float ar = er * c, ai = er * s;             // Lambda_bar
  abr[p] = ar; abi[p] = ai;
  float erc = expf(zr * (float)CHUNK);        // Lambda_bar^CHUNK = exp(z*CHUNK)
  float sc, cc;
  sincosf(zi * (float)CHUNK, &sc, &cc);
  apr[p] = erc * cc; api[p] = erc * sc;
  float nr = ar - 1.0f, ni = ai;              // (Lambda_bar - 1)/Lambda
  float den = lr * lr + li * li;
  cfr[p] = (nr * lr + ni * li) / den;
  cfi[p] = (ni * lr - nr * li) / den;
}

// ---------------- B_bar = coeff * B  ->  bf16 (P x H) ----------------
__global__ __launch_bounds__(256) void s5_make_w(
    const float* __restrict__ bre, const float* __restrict__ bim,
    const float* __restrict__ cfr, const float* __restrict__ cfi,
    unsigned short* __restrict__ wre, unsigned short* __restrict__ wim) {
  int p = blockIdx.x;
  float cr = cfr[p], ci = cfi[p];
  for (int h = threadIdx.x; h < HDIM; h += blockDim.x) {
    size_t idx = (size_t)p * HDIM + h;
    float br = bre[idx], bi = bim[idx];
    wre[idx] = f2bf(cr * br - ci * bi);
    wim[idx] = f2bf(cr * bi + ci * br);
  }
}

// ---------------- x -> bf16 ----------------
__global__ __launch_bounds__(256) void s5_conv_x(const float* __restrict__ x,
                                                 unsigned short* __restrict__ xb) {
  size_t idx = (size_t)blockIdx.x * blockDim.x + threadIdx.x;
  if (idx < (size_t)LDIM * HDIM) xb[idx] = f2bf(x[idx]);
}

// ---------------- C -> bf16 with readout constants folded in ----------------
__global__ __launch_bounds__(256) void s5_conv_c(const float* __restrict__ cre,
                                                 const float* __restrict__ cim,
                                                 unsigned short* __restrict__ crb,
                                                 unsigned short* __restrict__ cib) {
  size_t idx = (size_t)blockIdx.x * blockDim.x + threadIdx.x;
  if (idx < (size_t)HDIM * PDIM) {
    crb[idx] = f2bf(2.0f * cre[idx]);
    cib[idx] = f2bf(-2.0f * cim[idx]);
  }
}

// ---------------- GEMM1: Bu = x @ B_bar^T (re & im fused) ----------------
__global__ __launch_bounds__(256) void s5_gemm_bu(
    const unsigned short* __restrict__ xb,    // L x H bf16
    const unsigned short* __restrict__ wre,   // P x H bf16
    const unsigned short* __restrict__ wim,   // P x H bf16
    float* __restrict__ bu_re, float* __restrict__ bu_im) {  // L x P f32
  __shared__ __align__(16) unsigned short As[2][128 * 32];
  __shared__ __align__(16) unsigned short Bsr[2][64 * 32];
  __shared__ __align__(16) unsigned short Bsi[2][64 * 32];

  const int tid = threadIdx.x;
  const int lane = tid & 31;
  const int wave = tid >> 5;
  const int waveM = wave & 3;   // 4 waves along M (rows of x)
  const int waveN = wave >> 2;  // 2 waves along N (states p)
  const int row0 = blockIdx.y * 128;
  const int col0 = blockIdx.x * 64;

  const v8f vzero = {0.f, 0.f, 0.f, 0.f, 0.f, 0.f, 0.f, 0.f};
  v8f accR[2][2], accI[2][2];
#pragma unroll
  for (int i = 0; i < 2; ++i)
#pragma unroll
    for (int j = 0; j < 2; ++j) { accR[i][j] = vzero; accI[i][j] = vzero; }

  const int arow = tid >> 1;           // 0..127
  const int akoff = (tid & 1) * 16;
  const int brow = tid >> 2;           // 0..63
  const int bkoff = (tid & 3) * 8;

  const unsigned short* gA  = xb  + (size_t)(row0 + arow) * HDIM + akoff;
  const unsigned short* gBr = wre + (size_t)(col0 + brow) * HDIM + bkoff;
  const unsigned short* gBi = wim + (size_t)(col0 + brow) * HDIM + bkoff;

  auto stage = [&](int kk, int buf) {
    async_b128(gA + kk,     &As[buf][arow * 32 + akoff]);
    async_b128(gA + kk + 8, &As[buf][arow * 32 + akoff + 8]);
    async_b128(gBr + kk,    &Bsr[buf][brow * 32 + bkoff]);
    async_b128(gBi + kk,    &Bsi[buf][brow * 32 + bkoff]);
  };

  constexpr int NK = HDIM / 32;
  stage(0, 0);
  for (int it = 0; it < NK; ++it) {
    const int buf = it & 1;
    if (it + 1 < NK) {
      stage((it + 1) * 32, buf ^ 1);      // overlap next-tile DMA with compute
      if (it + 2 < NK) {
        __builtin_prefetch(gA + (it + 2) * 32, 0, 3);
        __builtin_prefetch(gBr + (it + 2) * 32, 0, 3);
        __builtin_prefetch(gBi + (it + 2) * 32, 0, 3);
      }
      wait_async<4>();                    // current tile done; 4 next-tile in flight
    } else {
      wait_async<0>();
    }
    __syncthreads();

    const int r16 = lane & 15, khigh = lane >> 4;
    Frag a[2], fr[2], fi[2];
#pragma unroll
    for (int mi = 0; mi < 2; ++mi) load_frag_a(As[buf], waveM * 32 + mi * 16 + r16, khigh, a[mi]);
#pragma unroll
    for (int ni = 0; ni < 2; ++ni) {
      load_frag_b(Bsr[buf], waveN * 32 + ni * 16 + r16, khigh, fr[ni]);
      load_frag_b(Bsi[buf], waveN * 32 + ni * 16 + r16, khigh, fi[ni]);
    }
#pragma unroll
    for (int mi = 0; mi < 2; ++mi)
#pragma unroll
      for (int ni = 0; ni < 2; ++ni) {
        accR[mi][ni] = __builtin_amdgcn_wmma_f32_16x16x32_bf16(
            false, a[mi].v, false, fr[ni].v, (short)0, accR[mi][ni], false, false);
        accI[mi][ni] = __builtin_amdgcn_wmma_f32_16x16x32_bf16(
            false, a[mi].v, false, fi[ni].v, (short)0, accI[mi][ni], false, false);
      }
    __syncthreads();
  }

  const int r16 = lane & 15;
  const int mh = (lane >> 4) * 8;   // C/D layout: lanes 16-31 hold M=r+8
#pragma unroll
  for (int mi = 0; mi < 2; ++mi)
#pragma unroll
    for (int ni = 0; ni < 2; ++ni) {
      int n = col0 + waveN * 32 + ni * 16 + r16;
#pragma unroll
      for (int r = 0; r < 8; ++r) {
        int m = row0 + waveM * 32 + mi * 16 + mh + r;
        bu_re[(size_t)m * PDIM + n] = accR[mi][ni][r];
        bu_im[(size_t)m * PDIM + n] = accI[mi][ni][r];
      }
    }
}

// ---------------- chunked parallel scan: local pass ----------------
__global__ __launch_bounds__(256) void s5_scan_local(
    const float* __restrict__ abr, const float* __restrict__ abi,
    float* __restrict__ bu_re, float* __restrict__ bu_im,
    float* __restrict__ car, float* __restrict__ cai) {
  int p = blockIdx.x * 256 + threadIdx.x;
  int chunk = blockIdx.y;
  float ar = abr[p], ai = abi[p];
  float hr = 0.f, hi = 0.f;
  size_t base = (size_t)chunk * CHUNK * PDIM + p;
  for (int t = 0; t < CHUNK; ++t) {
    size_t idx = base + (size_t)t * PDIM;
    float ur = bu_re[idx], ui = bu_im[idx];
    float nr = ar * hr - ai * hi + ur;
    float ni = ar * hi + ai * hr + ui;
    hr = nr; hi = ni;
    bu_re[idx] = hr; bu_im[idx] = hi;   // in-place local scan
  }
  car[chunk * PDIM + p] = hr;
  cai[chunk * PDIM + p] = hi;
}

// ---------------- scan: combine chunk carries (64 steps, per p) ----------------
__global__ __launch_bounds__(256) void s5_scan_prefix(
    const float* __restrict__ apr, const float* __restrict__ api,
    const float* __restrict__ car, const float* __restrict__ cai,
    float* __restrict__ prer, float* __restrict__ prei) {
  int p = blockIdx.x * 256 + threadIdx.x;
  float ar = apr[p], ai = api[p];     // Lambda_bar^CHUNK
  float sr = 0.f, si = 0.f;
  for (int c = 0; c < NCHUNK; ++c) {
    prer[c * PDIM + p] = sr;
    prei[c * PDIM + p] = si;
    float cr = car[c * PDIM + p], ci = cai[c * PDIM + p];
    float nr = ar * sr - ai * si + cr;
    float ni = ar * si + ai * sr + ci;
    sr = nr; si = ni;
  }
}

// ---------------- scan: apply prefix + convert xs to bf16 ----------------
__global__ __launch_bounds__(256) void s5_scan_apply(
    const float* __restrict__ abr, const float* __restrict__ abi,
    const float* __restrict__ prer, const float* __restrict__ prei,
    const float* __restrict__ bu_re, const float* __restrict__ bu_im,
    unsigned short* __restrict__ xsre, unsigned short* __restrict__ xsim) {
  int p = blockIdx.x * 256 + threadIdx.x;
  int chunk = blockIdx.y;
  float ar = abr[p], ai = abi[p];
  float sr = prer[chunk * PDIM + p], si = prei[chunk * PDIM + p];
  size_t base = (size_t)chunk * CHUNK * PDIM + p;
  for (int t = 0; t < CHUNK; ++t) {
    float nr = ar * sr - ai * si;   // A^(t+1) * prefix
    float ni = ar * si + ai * sr;
    sr = nr; si = ni;
    size_t idx = base + (size_t)t * PDIM;
    xsre[idx] = f2bf(bu_re[idx] + sr);
    xsim[idx] = f2bf(bu_im[idx] + si);
  }
}

// ---------------- GEMM2: y = xs_re @ (2C_re)^T + xs_im @ (-2C_im)^T ----------------
__global__ __launch_bounds__(256) void s5_gemm_y(
    const unsigned short* __restrict__ xsre,  // L x P bf16
    const unsigned short* __restrict__ xsim,  // L x P bf16
    const unsigned short* __restrict__ crb,   // H x P bf16
    const unsigned short* __restrict__ cib,   // H x P bf16
    float* __restrict__ y) {                  // L x H f32
  __shared__ __align__(16) unsigned short Asr[2][128 * 32];
  __shared__ __align__(16) unsigned short Asi[2][128 * 32];
  __shared__ __align__(16) unsigned short Bsr[2][64 * 32];
  __shared__ __align__(16) unsigned short Bsi[2][64 * 32];

  const int tid = threadIdx.x;
  const int lane = tid & 31;
  const int wave = tid >> 5;
  const int waveM = wave & 3;
  const int waveN = wave >> 2;
  const int row0 = blockIdx.y * 128;
  const int col0 = blockIdx.x * 64;

  const v8f vzero = {0.f, 0.f, 0.f, 0.f, 0.f, 0.f, 0.f, 0.f};
  v8f acc[2][2];
#pragma unroll
  for (int i = 0; i < 2; ++i)
#pragma unroll
    for (int j = 0; j < 2; ++j) acc[i][j] = vzero;

  const int arow = tid >> 1;
  const int akoff = (tid & 1) * 16;
  const int brow = tid >> 2;
  const int bkoff = (tid & 3) * 8;

  const unsigned short* gAr = xsre + (size_t)(row0 + arow) * PDIM + akoff;
  const unsigned short* gAi = xsim + (size_t)(row0 + arow) * PDIM + akoff;
  const unsigned short* gBr = crb + (size_t)(col0 + brow) * PDIM + bkoff;
  const unsigned short* gBi = cib + (size_t)(col0 + brow) * PDIM + bkoff;

  auto stage = [&](int kk, int buf) {
    async_b128(gAr + kk,     &Asr[buf][arow * 32 + akoff]);
    async_b128(gAr + kk + 8, &Asr[buf][arow * 32 + akoff + 8]);
    async_b128(gAi + kk,     &Asi[buf][arow * 32 + akoff]);
    async_b128(gAi + kk + 8, &Asi[buf][arow * 32 + akoff + 8]);
    async_b128(gBr + kk,     &Bsr[buf][brow * 32 + bkoff]);
    async_b128(gBi + kk,     &Bsi[buf][brow * 32 + bkoff]);
  };

  constexpr int NK = PDIM / 32;
  stage(0, 0);
  for (int it = 0; it < NK; ++it) {
    const int buf = it & 1;
    if (it + 1 < NK) {
      stage((it + 1) * 32, buf ^ 1);
      if (it + 2 < NK) {
        __builtin_prefetch(gAr + (it + 2) * 32, 0, 3);
        __builtin_prefetch(gAi + (it + 2) * 32, 0, 3);
      }
      wait_async<6>();                    // tolerate the 6 next-tile requests
    } else {
      wait_async<0>();
    }
    __syncthreads();

    const int r16 = lane & 15, khigh = lane >> 4;
    Frag are[2], aim[2], fr[2], fi[2];
#pragma unroll
    for (int mi = 0; mi < 2; ++mi) {
      load_frag_a(Asr[buf], waveM * 32 + mi * 16 + r16, khigh, are[mi]);
      load_frag_a(Asi[buf], waveM * 32 + mi * 16 + r16, khigh, aim[mi]);
    }
#pragma unroll
    for (int ni = 0; ni < 2; ++ni) {
      load_frag_b(Bsr[buf], waveN * 32 + ni * 16 + r16, khigh, fr[ni]);
      load_frag_b(Bsi[buf], waveN * 32 + ni * 16 + r16, khigh, fi[ni]);
    }
#pragma unroll
    for (int mi = 0; mi < 2; ++mi)
#pragma unroll
      for (int ni = 0; ni < 2; ++ni) {
        acc[mi][ni] = __builtin_amdgcn_wmma_f32_16x16x32_bf16(
            false, are[mi].v, false, fr[ni].v, (short)0, acc[mi][ni], false, false);
        acc[mi][ni] = __builtin_amdgcn_wmma_f32_16x16x32_bf16(
            false, aim[mi].v, false, fi[ni].v, (short)0, acc[mi][ni], false, false);
      }
    __syncthreads();
  }

  const int r16 = lane & 15;
  const int mh = (lane >> 4) * 8;
#pragma unroll
  for (int mi = 0; mi < 2; ++mi)
#pragma unroll
    for (int ni = 0; ni < 2; ++ni) {
      int n = col0 + waveN * 32 + ni * 16 + r16;
#pragma unroll
      for (int r = 0; r < 8; ++r) {
        int m = row0 + waveM * 32 + mi * 16 + mh + r;
        y[(size_t)m * HDIM + n] = acc[mi][ni][r];
      }
    }
}

// ---------------- residual + LayerNorm ----------------
__global__ __launch_bounds__(256) void s5_ln(
    const float* __restrict__ x, const float* __restrict__ y,
    const float* __restrict__ dvec, const float* __restrict__ gamma,
    const float* __restrict__ beta, float* __restrict__ out) {
  __shared__ float s1[256];
  __shared__ float s2[256];
  int l = blockIdx.x;
  const float* xr = x + (size_t)l * HDIM;
  const float* yr = y + (size_t)l * HDIM;
  float vals[4];
  float sum = 0.f, sq = 0.f;
#pragma unroll
  for (int i = 0; i < 4; ++i) {
    int h = threadIdx.x + i * 256;
    float hv = xr[h] * (1.0f + dvec[h]) + yr[h];   // skip + x*D + 2Re(xs C^T)
    vals[i] = hv;
    sum += hv; sq += hv * hv;
  }
  s1[threadIdx.x] = sum; s2[threadIdx.x] = sq;
  __syncthreads();
  for (int off = 128; off > 0; off >>= 1) {
    if (threadIdx.x < off) {
      s1[threadIdx.x] += s1[threadIdx.x + off];
      s2[threadIdx.x] += s2[threadIdx.x + off];
    }
    __syncthreads();
  }
  float mean = s1[0] * (1.0f / HDIM);
  float var = s2[0] * (1.0f / HDIM) - mean * mean;
  float inv = rsqrtf(var + EPS);
#pragma unroll
  for (int i = 0; i < 4; ++i) {
    int h = threadIdx.x + i * 256;
    out[(size_t)l * HDIM + h] = (vals[i] - mean) * inv * gamma[h] + beta[h];
  }
}

extern "C" void kernel_launch(void* const* d_in, const int* in_sizes, int n_in,
                              void* d_out, int out_size, void* d_ws, size_t ws_size,
                              hipStream_t stream) {
  const float* x     = (const float*)d_in[0];
  const float* lre   = (const float*)d_in[1];
  const float* lim   = (const float*)d_in[2];
  const float* bre   = (const float*)d_in[3];
  const float* bim   = (const float*)d_in[4];
  const float* cre   = (const float*)d_in[5];
  const float* cim   = (const float*)d_in[6];
  const float* dvec  = (const float*)d_in[7];
  const float* logdt = (const float*)d_in[8];
  const float* gamma = (const float*)d_in[9];
  const float* beta  = (const float*)d_in[10];
  float* out = (float*)d_out;

  char* w = (char*)d_ws;
  auto take = [&](size_t bytes) -> void* {
    void* p = (void*)w;
    w += (bytes + 255) & ~(size_t)255;
    return p;
  };
  unsigned short* xb   = (unsigned short*)take((size_t)LDIM * HDIM * 2);
  unsigned short* wre  = (unsigned short*)take((size_t)PDIM * HDIM * 2);
  unsigned short* wim  = (unsigned short*)take((size_t)PDIM * HDIM * 2);
  unsigned short* crb  = (unsigned short*)take((size_t)HDIM * PDIM * 2);
  unsigned short* cib  = (unsigned short*)take((size_t)HDIM * PDIM * 2);
  float* bu_re = (float*)take((size_t)LDIM * PDIM * 4);
  float* bu_im = (float*)take((size_t)LDIM * PDIM * 4);
  unsigned short* xsre = (unsigned short*)take((size_t)LDIM * PDIM * 2);
  unsigned short* xsim = (unsigned short*)take((size_t)LDIM * PDIM * 2);
  float* ybuf = (float*)take((size_t)LDIM * HDIM * 4);
  float* abr = (float*)take(PDIM * 4);
  float* abi = (float*)take(PDIM * 4);
  float* apr = (float*)take(PDIM * 4);
  float* api = (float*)take(PDIM * 4);
  float* cfr = (float*)take(PDIM * 4);
  float* cfi = (float*)take(PDIM * 4);
  float* car = (float*)take((size_t)NCHUNK * PDIM * 4);
  float* cai = (float*)take((size_t)NCHUNK * PDIM * 4);
  float* prer = (float*)take((size_t)NCHUNK * PDIM * 4);
  float* prei = (float*)take((size_t)NCHUNK * PDIM * 4);

  // discretization scalars
  s5_setup<<<dim3(PDIM / 256), dim3(256), 0, stream>>>(lre, lim, logdt, abr, abi, apr, api, cfr, cfi);
  // bf16 operand prep
  s5_make_w<<<dim3(PDIM), dim3(256), 0, stream>>>(bre, bim, cfr, cfi, wre, wim);
  s5_conv_x<<<dim3((LDIM * HDIM) / 256), dim3(256), 0, stream>>>(x, xb);
  s5_conv_c<<<dim3((HDIM * PDIM) / 256), dim3(256), 0, stream>>>(cre, cim, crb, cib);
  // GEMM1: Bu = x @ B_bar^T (re/im)
  s5_gemm_bu<<<dim3(PDIM / 64, LDIM / 128), dim3(256), 0, stream>>>(xb, wre, wim, bu_re, bu_im);
  // chunked parallel scan along L
  s5_scan_local<<<dim3(PDIM / 256, NCHUNK), dim3(256), 0, stream>>>(abr, abi, bu_re, bu_im, car, cai);
  s5_scan_prefix<<<dim3(PDIM / 256), dim3(256), 0, stream>>>(apr, api, car, cai, prer, prei);
  s5_scan_apply<<<dim3(PDIM / 256, NCHUNK), dim3(256), 0, stream>>>(abr, abi, prer, prei, bu_re, bu_im, xsre, xsim);
  // GEMM2: y = 2*Re(xs @ C^T)
  s5_gemm_y<<<dim3(HDIM / 64, LDIM / 128), dim3(256), 0, stream>>>(xsre, xsim, crb, cib, ybuf);
  // residual + LayerNorm
  s5_ln<<<dim3(LDIM), dim3(256), 0, stream>>>(x, ybuf, dvec, gamma, beta, out);
}
